// EncConvNet_77189152244003
// MI455X (gfx1250) — matrix operand order repaired and checked by reading, not compiled, not run on hardware
//
#include <hip/hip_runtime.h>

// CDNA5 / gfx1250 fused EncConvNet:
//   conv(7x7,s3) + bias + square  -> LDS (f16)
//   fc1 (WMMA f16 16x16x32, K=256) + bias + square -> LDS (f16)
//   fc2 (WMMA f16 16x16x32, K=64 padded N=16) + bias -> global f32
// One block = 256 threads (8 wave32) = 128 samples. Grid = 65536/128 = 512.

typedef _Float16 h2   __attribute__((ext_vector_type(2)));
typedef _Float16 v8h  __attribute__((ext_vector_type(8)));
typedef _Float16 v16h __attribute__((ext_vector_type(16)));
typedef float    v8f  __attribute__((ext_vector_type(8)));

#define THREADS 256
#define BLK_SAMPLES 128

__global__ __launch_bounds__(THREADS)
void encconv_fused(const float* __restrict__ x,
                   const float* __restrict__ conv_w,
                   const float* __restrict__ conv_b,
                   const float* __restrict__ fc1_w,
                   const float* __restrict__ fc1_b,
                   const float* __restrict__ fc2_w,
                   const float* __restrict__ fc2_b,
                   float* __restrict__ out)
{
    // ---- LDS layout (~117 KB) ----
    __shared__ _Float16 sW1[64 * 256];          // fc1 weights, [n][k] f16 (32 KB)
    __shared__ _Float16 sW2[16 * 64];           // fc2 weights, zero-padded to N=16 (2 KB)
    __shared__ _Float16 sH1[BLK_SAMPLES * 256]; // squared conv activations (64 KB)
    __shared__ _Float16 sH2[8][16 * 64];        // per-wave squared fc1 activations (16 KB)
    __shared__ _Float16 sCW[49 * 4];            // conv weights packed [ij][c] for pk_fma

    const int tid = threadIdx.x;

    // ---- Phase 0: stage weights into LDS (f32 -> f16) ----
    for (int i = tid; i < 64 * 256; i += THREADS)
        sW1[i] = (_Float16)fc1_w[i];                       // fc1_w already [o][i] = [n][k]
    for (int i = tid; i < 16 * 64; i += THREADS) {
        const int n = i >> 6, k = i & 63;
        sW2[i] = (n < 10) ? (_Float16)fc2_w[n * 64 + k] : (_Float16)0.0f;
    }
    for (int i = tid; i < 49 * 4; i += THREADS) {
        const int ij = i >> 2, c = i & 3;                  // pairs (c0,c1),(c2,c3) adjacent
        sCW[i] = (_Float16)conv_w[c * 49 + ij];
    }
    __syncthreads();

    // ---- Phase 1: conv + bias + square, all 4 channels per thread ----
    const float cb0 = conv_b[0], cb1 = conv_b[1], cb2 = conv_b[2], cb3 = conv_b[3];
    const int posId = tid & 63;          // output position 0..63 (r*8+cc)
    const int sSub  = tid >> 6;          // 0..3 samples in flight per round
    const int r  = posId >> 3;
    const int cc = posId & 7;
    const int wbase = (3 * r) * 28 + 3 * cc;
    const float* xblk = x + (size_t)blockIdx.x * (BLK_SAMPLES * 784);

    for (int chunk = 0; chunk < BLK_SAMPLES / 4; ++chunk) {
        const int sl = chunk * 4 + sSub;                   // 0..127
        const float* xs = xblk + sl * 784;
        if (sl + 4 < BLK_SAMPLES)
            __builtin_prefetch((const void*)(xs + 4 * 784), 0, 1); // global_prefetch_b8

        h2 acc01 = { (_Float16)0.0f, (_Float16)0.0f };
        h2 acc23 = acc01;
#pragma unroll
        for (int i = 0; i < 7; ++i) {
#pragma unroll
            for (int j = 0; j < 7; ++j) {
                const _Float16 xh = (_Float16)xs[wbase + i * 28 + j];
                const h2 xx = { xh, xh };
                const h2 w01 = *(const h2*)&sCW[(i * 7 + j) * 4];
                const h2 w23 = *(const h2*)&sCW[(i * 7 + j) * 4 + 2];
                acc01 += xx * w01;                          // v_pk_fma_f16
                acc23 += xx * w23;
            }
        }
        float t;
        t = (float)acc01[0] + cb0; sH1[sl * 256 +   0 + posId] = (_Float16)(t * t);
        t = (float)acc01[1] + cb1; sH1[sl * 256 +  64 + posId] = (_Float16)(t * t);
        t = (float)acc23[0] + cb2; sH1[sl * 256 + 128 + posId] = (_Float16)(t * t);
        t = (float)acc23[1] + cb3; sH1[sl * 256 + 192 + posId] = (_Float16)(t * t);
    }
    __syncthreads();

    // ---- Phase 2: fc1 via WMMA, one 16-sample M-tile per wave ----
    const int wave = tid >> 5;           // warpSize == 32 on gfx1250
    const int lane = tid & 31;
    const int ln   = lane & 15;          // A: M row / B: N col / D: N col
    const int hi   = lane >> 4;          // half-wave selector per ISA layouts

    union AB { v16h v; v8h h[2]; };

    v8f acc0 = {}, acc1 = {}, acc2n = {}, acc3 = {};
    const _Float16* aBase = &sH1[wave * 16 * 256];
#pragma unroll
    for (int kk = 0; kk < 8; ++kk) {
        const int k0 = kk * 32;
        AB a;
        // 16-bit A 16x32 layout: lane ln holds M=ln; K chunks k0+hi*8 and k0+16+hi*8
        a.h[0] = *(const v8h*)&aBase[ln * 256 + k0 + hi * 8];
        a.h[1] = *(const v8h*)&aBase[ln * 256 + k0 + 16 + hi * 8];
#pragma unroll
        for (int nt = 0; nt < 4; ++nt) {
            AB b;
            // 16-bit B 32x16 layout: lane ln holds N=ln; K chunks k0+hi*8 and k0+16+hi*8
            const _Float16* bp = &sW1[(nt * 16 + ln) * 256 + k0 + hi * 8];
            b.h[0] = *(const v8h*)bp;
            b.h[1] = *(const v8h*)(bp + 16);
            v8f& acc = (nt == 0) ? acc0 : (nt == 1) ? acc1 : (nt == 2) ? acc2n : acc3;
            acc = __builtin_amdgcn_wmma_f32_16x16x32_f16(
                      false, a.v, false, b.v, (short)0, acc, false, false);
        }
    }

    // bias + square -> per-wave f16 A-tile for fc2
#pragma unroll
    for (int nt = 0; nt < 4; ++nt) {
        const v8f& acc = (nt == 0) ? acc0 : (nt == 1) ? acc1 : (nt == 2) ? acc2n : acc3;
        const float b1 = fc1_b[nt * 16 + ln];
#pragma unroll
        for (int r2 = 0; r2 < 8; ++r2) {
            const float v = acc[r2] + b1;                  // D row = r2 + hi*8, col = ln
            sH2[wave][(r2 + hi * 8) * 64 + nt * 16 + ln] = (_Float16)(v * v);
        }
    }
    __syncthreads();

    // ---- Phase 3: fc2 via WMMA (K=64 -> 2 steps), N padded 10->16 ----
    v8f accO = {};
    const _Float16* a2 = &sH2[wave][0];
#pragma unroll
    for (int kk = 0; kk < 2; ++kk) {
        const int k0 = kk * 32;
        AB a, b;
        a.h[0] = *(const v8h*)&a2[ln * 64 + k0 + hi * 8];
        a.h[1] = *(const v8h*)&a2[ln * 64 + k0 + 16 + hi * 8];
        b.h[0] = *(const v8h*)&sW2[ln * 64 + k0 + hi * 8];
        b.h[1] = *(const v8h*)&sW2[ln * 64 + k0 + 16 + hi * 8];
        accO = __builtin_amdgcn_wmma_f32_16x16x32_f16(
                   false, a.v, false, b.v, (short)0, accO, false, false);
    }

    // ---- Output: D layout -> out[row*10 + col], cols 0..9 valid ----
    if (ln < 10) {
        const float b2 = fc2_b[ln];
        const size_t rowbase = (size_t)blockIdx.x * BLK_SAMPLES + wave * 16 + hi * 8;
#pragma unroll
        for (int r2 = 0; r2 < 8; ++r2)
            out[(rowbase + r2) * 10 + ln] = accO[r2] + b2;
    }
}

extern "C" void kernel_launch(void* const* d_in, const int* in_sizes, int n_in,
                              void* d_out, int out_size, void* d_ws, size_t ws_size,
                              hipStream_t stream) {
    (void)n_in; (void)out_size; (void)d_ws; (void)ws_size;
    const float* x      = (const float*)d_in[0];
    const float* conv_w = (const float*)d_in[1];
    const float* conv_b = (const float*)d_in[2];
    const float* fc1_w  = (const float*)d_in[3];
    const float* fc1_b  = (const float*)d_in[4];
    const float* fc2_w  = (const float*)d_in[5];
    const float* fc2_b  = (const float*)d_in[6];
    float* out = (float*)d_out;

    const int B = in_sizes[0] / 784;           // 65536
    dim3 grid(B / BLK_SAMPLES);                // 512 blocks
    encconv_fused<<<grid, THREADS, 0, stream>>>(x, conv_w, conv_b,
                                                fc1_w, fc1_b, fc2_w, fc2_b, out);
}
